// LinkNet_9706626089664
// MI455X (gfx1250) — matrix operand order repaired and checked by reading, not compiled, not run on hardware
//
#include <hip/hip_runtime.h>

// ---------------------------------------------------------------------------
// LinkNet (TransformerConv GNN) for MI455X / gfx1250.
// All GEMMs on v_wmma_f32_16x16x32_f16 (f16 in, f32 accum), fully K-unrolled.
// Scatter/softmax/layernorm in f32 VALU with atomics.
//
// Assumed d_in flattening (depth-first, dict insertion order):
//  0 x[N,6] 1 edge_attr[E,4] 2 edge_index[2,E](int)
//  3 W0n[6,64] 4 b0n 5 W1n[64,64] 6 b1n            (node_emb)
//  7 W0e[4,64] 8 b0e 9 W1e[64,64] 10 b1e           (edge_emb)
//  11 g_comb[128] 12 bb_comb[128]                  (norm_comb)
//  13 Wq[128,256] 14 bq 15 Wk 16 bk 17 Wv 18 bv 19 Wskip 20 bskip
//  21 g_tc[256] 22 bb_tc[256]                      (norm_tc)
//  23 Wn[256,64] 24 bn   25 We[256,64] 26 be       (proj_node / proj_edge)
//  27 Wc1[128,64] 28 bc1 29 Wc2[64,1] 30 bc2[1]    (clf)
// ---------------------------------------------------------------------------

#define NNODES 50000
#define NEDGES 150000
#define E2     (2 * NEDGES)
#define HID    64

typedef __attribute__((ext_vector_type(16))) _Float16 v16h;
typedef __attribute__((ext_vector_type(8)))  float    v8f;
typedef __attribute__((ext_vector_type(2)))  _Float16 h2;
typedef __attribute__((ext_vector_type(2)))  float    f2;

__device__ __forceinline__ void atomicMaxF(float* a, float v) {
  // sign-aware monotonic-bits trick; location initialized to -inf bits.
  if (v >= 0.f) atomicMax((int*)a, __float_as_int(v));
  else          atomicMin((unsigned int*)a, __float_as_uint(v));
}

// ----------------------- WMMA GEMM: C[M,NC] = A[M,K] @ W[K,NC] + b ----------
// MODE 0: A16 = f16 row-major [M,K]
// MODE 1: A row m = concat(h[src0[m]], h[dst0[m]])   (K==128, f32 gather)
// MODE 2: A row m = out16[send[m]] - out16[recv[m]]  (K==256, f16 gather)
// grid = (M/16, NC/64), block = 128 (4 waves, one 16x16 N-tile per wave).
template <int MODE, int K, int NC, bool RELU, bool RESID>
__global__ void __launch_bounds__(128)
k_gemm_wmma(const _Float16* __restrict__ A16,
            const float*    __restrict__ gsrc,   // MODE1: h
            const _Float16* __restrict__ g16,    // MODE2: out16
            const int*      __restrict__ ei,
            const float*    __restrict__ W,
            const float*    __restrict__ bias,
            const float*    __restrict__ resid,
            float*          __restrict__ outF)
{
  const int wave = threadIdx.x >> 5;
  const int lane = threadIdx.x & 31;
  const int half = lane >> 4;
  const int lr   = lane & 15;
  const int m0   = blockIdx.x * 16;
  const int n0   = blockIdx.y * 64 + wave * 16;
  const int row  = m0 + lr;             // A-tile row owned by this lane

  int ia = 0, ib = 0;
  if constexpr (MODE == 1) { ia = ei[row]; ib = ei[row + NEDGES]; }
  if constexpr (MODE == 2) {
    ia = ei[row];                                              // send
    ib = (row < NEDGES) ? ei[row + NEDGES] : ei[row - NEDGES]; // recv
  }

  v8f acc = {};
#pragma unroll
  for (int k0 = 0; k0 < K; k0 += 32) {
    v16h a, b;
#pragma unroll
    for (int j = 0; j < 8; ++j) {
      // A 16x32 f16 layout: half 0: j<4 -> K=2j, j>=4 -> K=16+2(j-4); half 1: +8
      const int kk = k0 + ((j < 4) ? (half * 8 + 2 * j)
                                   : (16 + half * 8 + 2 * (j - 4)));
      if constexpr (MODE == 0) {
        const h2 p = *(const h2*)(A16 + (size_t)row * K + kk);   // one b32 load
        a[2 * j] = p[0]; a[2 * j + 1] = p[1];
      } else if constexpr (MODE == 1) {
        const int rr = (kk < HID) ? ia : ib;   // kk even; pair stays in block
        const f2 p = *(const f2*)(gsrc + (size_t)rr * HID + (kk & (HID - 1)));
        a[2 * j] = (_Float16)p[0]; a[2 * j + 1] = (_Float16)p[1];
      } else {
        const h2 pa = *(const h2*)(g16 + (size_t)ia * 256 + kk);
        const h2 pb = *(const h2*)(g16 + (size_t)ib * 256 + kk);
        a[2 * j]     = (_Float16)((float)pa[0] - (float)pb[0]);
        a[2 * j + 1] = (_Float16)((float)pa[1] - (float)pb[1]);
      }
      // B 32x16 layout: lane col = lr, same K mapping as A (kk)
      const int kb = kk;
      b[2 * j]     = (_Float16)W[(size_t)kb * NC + n0 + lr];
      b[2 * j + 1] = (_Float16)W[(size_t)(kb + 1) * NC + n0 + lr];
    }
    acc = __builtin_amdgcn_wmma_f32_16x16x32_f16(false, a, false, b,
                                                 (short)0, acc, false, false);
  }
  // C layout: VGPR j, lane -> (M = j + 8*half, N = lr)
  const int    n    = n0 + lr;
  const float  bn   = bias[n];
  const size_t base = (size_t)(m0 + 8 * half) * NC + n;
#pragma unroll
  for (int j = 0; j < 8; ++j) {
    float v = acc[j] + bn;
    if constexpr (RELU)  v = fmaxf(v, 0.f);
    if constexpr (RESID) v += resid[base + (size_t)j * NC];
    outF[base + (size_t)j * NC] = v;
  }
}

// ----------------------- small elementwise / scatter kernels ----------------
__global__ void k_node_l1(const float* __restrict__ x, const float* __restrict__ W,
                          const float* __restrict__ b, _Float16* __restrict__ o) {
  int t = blockIdx.x * 256 + threadIdx.x;
  if (t >= NNODES * HID) return;
  int c = t & 63, n = t >> 6;
  float s = b[c];
#pragma unroll
  for (int i = 0; i < 6; ++i) s += x[n * 6 + i] * W[i * HID + c];
  o[t] = (_Float16)fmaxf(s, 0.f);
}

__global__ void k_edge_l1(const float* __restrict__ ea, const float* __restrict__ W,
                          const float* __restrict__ b, _Float16* __restrict__ o) {
  int t = blockIdx.x * 256 + threadIdx.x;
  if (t >= E2 * HID) return;
  int c = t & 63, e = t >> 6;
  int j = (e < NEDGES) ? e : e - NEDGES;   // e_bi = concat(edge_attr, edge_attr)
  float s = b[c];
#pragma unroll
  for (int i = 0; i < 4; ++i) s += ea[j * 4 + i] * W[i * HID + c];
  o[t] = (_Float16)fmaxf(s, 0.f);
}

__global__ void k_zero(float* p, int n) {
  int t = blockIdx.x * 256 + threadIdx.x;
  if (t < n) p[t] = 0.f;
}
__global__ void k_neginf(float* p, int n) {
  int t = blockIdx.x * 256 + threadIdx.x;
  if (t < n) ((int*)p)[t] = 0xFF800000;    // -inf
}

__global__ void k_agg(const float* __restrict__ e, const int* __restrict__ ei,
                      float* __restrict__ agg) {
  int t = blockIdx.x * 256 + threadIdx.x;
  if (t >= E2 * HID) return;
  int c = t & 63, e2 = t >> 6;
  int r = (e2 < NEDGES) ? ei[e2 + NEDGES] : ei[e2 - NEDGES];  // recv
  atomicAdd(&agg[(size_t)r * HID + c], e[t]);
}

// layer_norm over 128 chans of concat(h, agg-h); one wave per node.
__global__ void k_ln_comb(const float* __restrict__ h, const float* __restrict__ agg,
                          const float* __restrict__ g, const float* __restrict__ b,
                          _Float16* __restrict__ o) {
  int nid = blockIdx.x * 4 + (threadIdx.x >> 5);
  int lane = threadIdx.x & 31;
  if (nid >= NNODES) return;
  float v[4], s = 0.f, s2 = 0.f;
#pragma unroll
  for (int i = 0; i < 4; ++i) {
    int c = lane + 32 * i;
    float x;
    if (c < HID) x = h[(size_t)nid * HID + c];
    else { int cc = c - HID; x = agg[(size_t)nid * HID + cc] - h[(size_t)nid * HID + cc]; }
    v[i] = x; s += x; s2 += x * x;
  }
  for (int off = 16; off > 0; off >>= 1) {
    s += __shfl_xor(s, off, 32); s2 += __shfl_xor(s2, off, 32);
  }
  float m = s * (1.f / 128.f);
  float inv = rsqrtf(s2 * (1.f / 128.f) - m * m + 1e-5f);
#pragma unroll
  for (int i = 0; i < 4; ++i) {
    int c = lane + 32 * i;
    o[(size_t)nid * 128 + c] = (_Float16)((v[i] - m) * inv * g[c] + b[c]);
  }
}

// layer_norm over 256 chans of attention output; one wave per node; f16 out.
__global__ void k_ln_out(const float* __restrict__ x, const float* __restrict__ g,
                         const float* __restrict__ b, _Float16* __restrict__ o) {
  int nid = blockIdx.x * 4 + (threadIdx.x >> 5);
  int lane = threadIdx.x & 31;
  if (nid >= NNODES) return;
  float v[8], s = 0.f, s2 = 0.f;
#pragma unroll
  for (int i = 0; i < 8; ++i) {
    int c = lane + 32 * i;
    float t = x[(size_t)nid * 256 + c];
    v[i] = t; s += t; s2 += t * t;
  }
  for (int off = 16; off > 0; off >>= 1) {
    s += __shfl_xor(s, off, 32); s2 += __shfl_xor(s2, off, 32);
  }
  float m = s * (1.f / 256.f);
  float inv = rsqrtf(s2 * (1.f / 256.f) - m * m + 1e-5f);
#pragma unroll
  for (int i = 0; i < 8; ++i) {
    int c = lane + 32 * i;
    o[(size_t)nid * 256 + c] = (_Float16)((v[i] - m) * inv * g[c] + b[c]);
  }
}

__global__ void k_logits(const float* __restrict__ q, const float* __restrict__ k,
                         const int* __restrict__ ei, float* __restrict__ exb,
                         float* __restrict__ mx) {
  int t = blockIdx.x * 256 + threadIdx.x;
  if (t >= E2 * 4) return;
  int hd = t & 3, e2 = t >> 2;
  int s = ei[e2];
  int r = (e2 < NEDGES) ? ei[e2 + NEDGES] : ei[e2 - NEDGES];
  const float* qp = q + (size_t)r * 256 + hd * 64;   // q[dst]
  const float* kp = k + (size_t)s * 256 + hd * 64;   // k[src]
  float acc = 0.f;
#pragma unroll
  for (int d = 0; d < 64; ++d) acc += qp[d] * kp[d];
  acc *= 0.125f;                                     // 1/sqrt(64)
  exb[t] = acc;
  atomicMaxF(&mx[(size_t)r * 4 + hd], acc);
}

__global__ void k_expn(const int* __restrict__ ei, float* __restrict__ exb,
                       const float* __restrict__ mx, float* __restrict__ dn) {
  int t = blockIdx.x * 256 + threadIdx.x;
  if (t >= E2 * 4) return;
  int hd = t & 3, e2 = t >> 2;
  int r = (e2 < NEDGES) ? ei[e2 + NEDGES] : ei[e2 - NEDGES];
  float ex = __expf(exb[t] - mx[(size_t)r * 4 + hd]);
  exb[t] = ex;
  atomicAdd(&dn[(size_t)r * 4 + hd], ex);
}

__global__ void k_scatter(const float* __restrict__ vv, const int* __restrict__ ei,
                          const float* __restrict__ exb, const float* __restrict__ dn,
                          float* __restrict__ out) {
  int t = blockIdx.x * 256 + threadIdx.x;
  if (t >= E2 * 256) return;
  int c = t & 255, e2 = t >> 8, hd = c >> 6;
  int s = ei[e2];
  int r = (e2 < NEDGES) ? ei[e2 + NEDGES] : ei[e2 - NEDGES];
  float al = exb[(size_t)e2 * 4 + hd] / fmaxf(dn[(size_t)r * 4 + hd], 1e-16f);
  atomicAdd(&out[(size_t)r * 256 + c], vv[(size_t)s * 256 + c] * al);
}

__global__ void k_clf2(const float* __restrict__ hid, const float* __restrict__ w2,
                       const float* __restrict__ b2, float* __restrict__ out) {
  int j = blockIdx.x * 256 + threadIdx.x;
  if (j >= NEDGES) return;
  float s = b2[0];
#pragma unroll
  for (int c = 0; c < 64; ++c) s += hid[(size_t)j * 64 + c] * w2[c];
  out[j] = s;
}

// ---------------------------------------------------------------------------
extern "C" void kernel_launch(void* const* d_in, const int* in_sizes, int n_in,
                              void* d_out, int out_size, void* d_ws, size_t ws_size,
                              hipStream_t stream) {
  (void)in_sizes; (void)n_in; (void)out_size; (void)ws_size;
  const float* x    = (const float*)d_in[0];
  const float* ea   = (const float*)d_in[1];
  const int*   ei   = (const int*)  d_in[2];
  const float* W0n  = (const float*)d_in[3];  const float* b0n = (const float*)d_in[4];
  const float* W1n  = (const float*)d_in[5];  const float* b1n = (const float*)d_in[6];
  const float* W0e  = (const float*)d_in[7];  const float* b0e = (const float*)d_in[8];
  const float* W1e  = (const float*)d_in[9];  const float* b1e = (const float*)d_in[10];
  const float* gC   = (const float*)d_in[11]; const float* bC  = (const float*)d_in[12];
  const float* Wq   = (const float*)d_in[13]; const float* bq  = (const float*)d_in[14];
  const float* Wk   = (const float*)d_in[15]; const float* bk  = (const float*)d_in[16];
  const float* Wv   = (const float*)d_in[17]; const float* bv  = (const float*)d_in[18];
  const float* Ws   = (const float*)d_in[19]; const float* bs  = (const float*)d_in[20];
  const float* gT   = (const float*)d_in[21]; const float* bT  = (const float*)d_in[22];
  const float* Wn   = (const float*)d_in[23]; const float* bn  = (const float*)d_in[24];
  const float* We   = (const float*)d_in[25]; const float* be  = (const float*)d_in[26];
  const float* Wc1  = (const float*)d_in[27]; const float* bc1 = (const float*)d_in[28];
  const float* Wc2  = (const float*)d_in[29]; const float* bc2 = (const float*)d_in[30];
  float* out = (float*)d_out;

  // workspace carve (floats)
  float* wsf = (float*)d_ws;
  size_t o = 0;
  float* h   = wsf + o; o += (size_t)NNODES * HID;
  float* e   = wsf + o; o += (size_t)E2 * HID;
  float* agg = wsf + o; o += (size_t)NNODES * HID;
  float* q   = wsf + o; o += (size_t)NNODES * 256;
  float* kB  = wsf + o; o += (size_t)NNODES * 256;
  float* vB  = wsf + o; o += (size_t)NNODES * 256;
  float* oB  = wsf + o; o += (size_t)NNODES * 256;
  float* exb = wsf + o; o += (size_t)E2 * 4;
  float* mx  = wsf + o; o += (size_t)NNODES * 4;
  float* dn  = wsf + o; o += (size_t)NNODES * 4;
  _Float16* s16 = (_Float16*)(wsf + o);   // shared f16 staging (max 2E*64 halves)
  float* hid = q;                         // classifier hidden reuses q

  const int B = 256;
  // node_emb: L1 (VALU) -> f16, L2 (WMMA) -> h
  k_node_l1<<<(NNODES * HID + B - 1) / B, B, 0, stream>>>(x, W0n, b0n, s16);
  k_gemm_wmma<0, 64, 64, true, false><<<dim3(NNODES / 16, 1), 128, 0, stream>>>(
      s16, nullptr, nullptr, nullptr, W1n, b1n, nullptr, h);
  // edge_emb: L1 -> f16, L2 (WMMA) -> e
  k_edge_l1<<<(E2 * HID + B - 1) / B, B, 0, stream>>>(ea, W0e, b0e, s16);
  k_gemm_wmma<0, 64, 64, true, false><<<dim3(E2 / 16, 1), 128, 0, stream>>>(
      s16, nullptr, nullptr, nullptr, W1e, b1e, nullptr, e);
  // agg = segment_sum(e, recv)
  k_zero<<<(NNODES * HID + B - 1) / B, B, 0, stream>>>(agg, NNODES * HID);
  k_agg<<<(E2 * HID + B - 1) / B, B, 0, stream>>>(e, ei, agg);
  // comb = LN(concat(h, agg-h)) -> f16
  k_ln_comb<<<(NNODES + 3) / 4, 128, 0, stream>>>(h, agg, gC, bC, s16);
  // q,k,v and skip (skip seeds the attention-output accumulator)
  k_gemm_wmma<0, 128, 256, false, false><<<dim3(NNODES / 16, 4), 128, 0, stream>>>(
      s16, nullptr, nullptr, nullptr, Wq, bq, nullptr, q);
  k_gemm_wmma<0, 128, 256, false, false><<<dim3(NNODES / 16, 4), 128, 0, stream>>>(
      s16, nullptr, nullptr, nullptr, Wk, bk, nullptr, kB);
  k_gemm_wmma<0, 128, 256, false, false><<<dim3(NNODES / 16, 4), 128, 0, stream>>>(
      s16, nullptr, nullptr, nullptr, Wv, bv, nullptr, vB);
  k_gemm_wmma<0, 128, 256, false, false><<<dim3(NNODES / 16, 4), 128, 0, stream>>>(
      s16, nullptr, nullptr, nullptr, Ws, bs, nullptr, oB);
  // segmented softmax over incoming edges + weighted scatter of v
  k_neginf<<<(NNODES * 4 + B - 1) / B, B, 0, stream>>>(mx, NNODES * 4);
  k_zero  <<<(NNODES * 4 + B - 1) / B, B, 0, stream>>>(dn, NNODES * 4);
  k_logits<<<(E2 * 4 + B - 1) / B, B, 0, stream>>>(q, kB, ei, exb, mx);
  k_expn  <<<(E2 * 4 + B - 1) / B, B, 0, stream>>>(ei, exb, mx, dn);
  k_scatter<<<(E2 * 256 + B - 1) / B, B, 0, stream>>>(vB, ei, exb, dn, oB);
  // LN(out) -> f16 staging
  k_ln_out<<<(NNODES + 3) / 4, 128, 0, stream>>>(oB, gT, bT, s16);
  // h = out@Wn + bn + h  (in place);  e = (out[send]-out[recv])@We + be + e
  k_gemm_wmma<0, 256, 64, false, true><<<dim3(NNODES / 16, 1), 128, 0, stream>>>(
      s16, nullptr, nullptr, nullptr, Wn, bn, h, h);
  k_gemm_wmma<2, 256, 64, false, true><<<dim3(E2 / 16, 1), 128, 0, stream>>>(
      nullptr, nullptr, s16, ei, We, be, e, e);
  // classifier: concat(h[src0],h[dst0]) @ Wc1 (+relu) via WMMA, then matvec
  k_gemm_wmma<1, 128, 64, true, false><<<dim3(NEDGES / 16, 1), 128, 0, stream>>>(
      nullptr, h, nullptr, ei, Wc1, bc1, nullptr, hid);
  k_clf2<<<(NEDGES + B - 1) / B, B, 0, stream>>>(hid, Wc2, bc2, out);
}
